// Similarity_Loss_SNN_31490700214354
// MI455X (gfx1250) — matrix operand order repaired and checked by reading, not compiled, not run on hardware
//
#include <hip/hip_runtime.h>
#include <hip/hip_bf16.h>

typedef __attribute__((ext_vector_type(16))) _Float16 v16h;
typedef __attribute__((ext_vector_type(8)))  _Float16 v8h;
typedef __attribute__((ext_vector_type(8)))  float    v8f;
typedef __attribute__((ext_vector_type(4)))  int      v4i;

#define BN    4096     // batch
#define DK    1024     // embedding dim
#define INV_T 20.0f    // 1 / TEMPERATURE

#define KT     32            // k per stage (WMMA f16 K)
#define LDA    40            // LDS row stride in halves (32 + 8 pad, 16B multiple)
#define ABUF   (128 * LDA)   // halves per LDS buffer (A and B both 128 x KT)
#define NSTAGE (DK / KT)     // 32 k-stages
#define NWN    4             // column groups per block (waves in N)

// gfx1250 async global->LDS path (ASYNCcnt). Fall back to sync staging if the
// toolchain doesn't declare the builtins.
#if defined(__has_builtin)
#if __has_builtin(__builtin_amdgcn_global_load_async_to_lds_b128) && \
    __has_builtin(__builtin_amdgcn_s_wait_asynccnt)
#define USE_ASYNC 1
#endif
#endif

#if defined(USE_ASYNC)
// Builtin signature (from hipcc diagnostic): b128 variant takes pointers to
// int4 in AS(1) (global) and AS(3) (LDS), plus imm offset and imm cpol.
typedef __attribute__((address_space(1))) v4i* GVp;   // global int4*
typedef __attribute__((address_space(3))) v4i* LVp;   // LDS int4*
#define ASYNC_WAIT(N) __builtin_amdgcn_s_wait_asynccnt(N)
#else
#define ASYNC_WAIT(N)
#endif

// ---------------------------------------------------------------------------
// Pass 1: fp32 row norms + exact fp32 diagonal sim, and f32 -> f16 conversion.
// ---------------------------------------------------------------------------
__global__ __launch_bounds__(256)
void prep_kernel(const float* __restrict__ txt, const float* __restrict__ img,
                 _Float16* __restrict__ Af16, _Float16* __restrict__ Bf16,
                 float* __restrict__ img_sq, float* __restrict__ txt_sq,
                 float* __restrict__ num) {
  const int i = blockIdx.x;
  const int t = threadIdx.x;
  float si = 0.f, st = 0.f, c = 0.f;
  for (int k = t; k < DK; k += 256) {
    float a = img[(size_t)i * DK + k];
    float b = txt[(size_t)i * DK + k];
    si += a * a;
    st += b * b;
    c  += a * b;
    Af16[(size_t)i * DK + k] = (_Float16)a;   // image rows -> A (M side)
    Bf16[(size_t)i * DK + k] = (_Float16)b;   // text rows  -> B (N side)
  }
  __shared__ float s0[256], s1[256], s2[256];
  s0[t] = si; s1[t] = st; s2[t] = c;
  __syncthreads();
  for (int off = 128; off > 0; off >>= 1) {
    if (t < off) { s0[t] += s0[t + off]; s1[t] += s1[t + off]; s2[t] += s2[t + off]; }
    __syncthreads();
  }
  if (t == 0) {
    img_sq[i] = s0[0];
    txt_sq[i] = s1[0];
    float sq = fmaxf(s0[0] + s1[0] - 2.f * s2[0], 0.f);
    num[i] = __expf(-__fsqrt_rn(sq) * INV_T);   // sim[i,i] in full fp32
  }
}

// Stage one 128xKT A chunk and one 128xKT B chunk into LDS (16B granules).
// 4 async (or sync b128) transfers per thread.
__device__ __forceinline__
void stage(const _Float16* __restrict__ gA, const _Float16* __restrict__ gB,
           _Float16* lA, _Float16* lB, int t) {
#pragma unroll
  for (int p = 0; p < 2; ++p) {
    const int g  = p * 256 + t;
    const int r  = g >> 2;
    const int kg = (g & 3) * 8;
#if defined(USE_ASYNC)
    __builtin_amdgcn_global_load_async_to_lds_b128(
        (GVp)(gA + (size_t)r * DK + kg), (LVp)(lA + r * LDA + kg), 0, 0);
    __builtin_amdgcn_global_load_async_to_lds_b128(
        (GVp)(gB + (size_t)r * DK + kg), (LVp)(lB + r * LDA + kg), 0, 0);
#else
    *(uint4*)(lA + r * LDA + kg) = *(const uint4*)(gA + (size_t)r * DK + kg);
    *(uint4*)(lB + r * LDA + kg) = *(const uint4*)(gB + (size_t)r * DK + kg);
#endif
  }
}

// ---------------------------------------------------------------------------
// Pass 2: WMMA f16 cross-GEMM fused with sqrt/exp epilogue and per-row
// partial-sum reduction. grid = (BN/128, 8), block = 256 (8 waves).
// Block tile 128x128 per j-step; wave tile 64x32 = 4x2 WMMA accumulators.
// Double-buffered LDS, async global->LDS staging.
// ---------------------------------------------------------------------------
__global__ __launch_bounds__(256)
void simrow_kernel(const _Float16* __restrict__ Af16,   // images [BN][DK]
                   const _Float16* __restrict__ Bf16,   // texts  [BN][DK]
                   const float* __restrict__ img_sq,
                   const float* __restrict__ txt_sq,
                   float* __restrict__ partial) {       // [8*NWN][BN]
  __shared__ _Float16 As[2 * ABUF];
  __shared__ _Float16 Bs[2 * ABUF];

  const int t      = threadIdx.x;
  const int lane   = t & 31;
  const int wave   = t >> 5;
  const int wm     = wave >> 2;           // 0..1 : 64-row group
  const int wn     = wave & 3;            // 0..3 : 32-col group
  const int iBase  = blockIdx.x * 128;
  const int jSlab  = blockIdx.y;          // 0..7, each covers 512 columns
  const int laneHi = (lane >= 16) ? 1 : 0;
  const int l16    = lane & 15;

  // Row squared-norms for the 32 row-slots this lane touches (constant in j).
  float rsq[4][8];
#pragma unroll
  for (int tm = 0; tm < 4; ++tm)
#pragma unroll
    for (int v = 0; v < 8; ++v)
      rsq[tm][v] = img_sq[iBase + wm * 64 + tm * 16 + v + laneHi * 8];

  float rowAcc[4][8];
#pragma unroll
  for (int tm = 0; tm < 4; ++tm)
#pragma unroll
    for (int v = 0; v < 8; ++v) rowAcc[tm][v] = 0.f;

  for (int jt = 0; jt < 4; ++jt) {
    const int jBase = jSlab * 512 + jt * 128;

    v8f acc[4][2];
#pragma unroll
    for (int tm = 0; tm < 4; ++tm)
#pragma unroll
      for (int tn = 0; tn < 2; ++tn) acc[tm][tn] = (v8f)0.f;

    const _Float16* gA = Af16 + (size_t)iBase * DK;
    const _Float16* gB = Bf16 + (size_t)jBase * DK;

    // Prologue: stage 0 into buffer 0.
    stage(gA, gB, As, Bs, t);

    for (int s = 0; s < NSTAGE; ++s) {
      const int cur = s & 1;
      if (s + 1 < NSTAGE) {
        // Issue next stage into the other buffer, then retire current stage
        // (our next 4 may stay in flight: ASYNCcnt <= 4).
        stage(gA + (s + 1) * KT, gB + (s + 1) * KT,
              As + (cur ^ 1) * ABUF, Bs + (cur ^ 1) * ABUF, t);
        ASYNC_WAIT(4);
      } else {
        ASYNC_WAIT(0);
      }
      __syncthreads();   // current buffer fully landed (all waves)

      const _Float16* A0 = As + cur * ABUF;
      const _Float16* B0 = Bs + cur * ABUF;

      // A fragment (ISA 7.12.2, 16-bit 16x32): lane<16 -> K 0..7,16..23 ;
      // lane>=16 -> K 8..15,24..31 (two contiguous 8-half runs).
      v16h af[4], bf[2];
#pragma unroll
      for (int tm = 0; tm < 4; ++tm) {
        const _Float16* p0 = A0 + (wm * 64 + tm * 16 + l16) * LDA + laneHi * 8;
        v8h lo = *(const v8h*)(p0);
        v8h hi = *(const v8h*)(p0 + 16);
#pragma unroll
        for (int e = 0; e < 8; ++e) { af[tm][e] = lo[e]; af[tm][e + 8] = hi[e]; }
      }
      // B fragment (32x16, K x N): lane<16 -> K 0..7,16..23 ; lane>=16 -> K 8..15,24..31.
#pragma unroll
      for (int tn = 0; tn < 2; ++tn) {
        const _Float16* p0 = B0 + (wn * 32 + tn * 16 + l16) * LDA + laneHi * 8;
        v8h lo = *(const v8h*)(p0);
        v8h hi = *(const v8h*)(p0 + 16);
#pragma unroll
        for (int e = 0; e < 8; ++e) { bf[tn][e] = lo[e]; bf[tn][e + 8] = hi[e]; }
      }

#pragma unroll
      for (int tm = 0; tm < 4; ++tm)
#pragma unroll
        for (int tn = 0; tn < 2; ++tn)
          acc[tm][tn] = __builtin_amdgcn_wmma_f32_16x16x32_f16(
              false, af[tm], false, bf[tn], (short)0, acc[tm][tn],
              false, false);

      __syncthreads();   // reads of this buffer done before it is re-staged
    }

    // Fused epilogue: sim = exp(-sqrt(max(||a||^2+||b||^2-2ab,0))/T),
    // accumulate per-row partial sums in registers.
#pragma unroll
    for (int tn = 0; tn < 2; ++tn) {
      float csq = txt_sq[jBase + wn * 32 + tn * 16 + l16];
#pragma unroll
      for (int tm = 0; tm < 4; ++tm)
#pragma unroll
        for (int v = 0; v < 8; ++v) {
          float sq = fmaxf(rsq[tm][v] + csq - 2.f * acc[tm][tn][v], 0.f);
          rowAcc[tm][v] += __expf(-__fsqrt_rn(sq) * INV_T);
        }
    }
  }

  // A row's 16 columns live across a 16-lane half (C-layout): reduce within
  // halves {0..15} and {16..31} independently (xor masks 1,2,4,8).
#pragma unroll
  for (int tm = 0; tm < 4; ++tm)
#pragma unroll
    for (int v = 0; v < 8; ++v) {
      float x = rowAcc[tm][v];
      x += __shfl_xor(x, 1, 32);
      x += __shfl_xor(x, 2, 32);
      x += __shfl_xor(x, 4, 32);
      x += __shfl_xor(x, 8, 32);
      rowAcc[tm][v] = x;
    }

  // Waves with equal wm share rows -> store into per-wn slabs (no race).
  if (l16 == 0) {
#pragma unroll
    for (int tm = 0; tm < 4; ++tm)
#pragma unroll
      for (int v = 0; v < 8; ++v) {
        int row = iBase + wm * 64 + tm * 16 + v + laneHi * 8;
        partial[((size_t)jSlab * NWN + wn) * BN + row] = rowAcc[tm][v];
      }
  }
}

// ---------------------------------------------------------------------------
// Pass 3: combine the 8*NWN partial slabs per row, per-row loss, scalar mean.
// ---------------------------------------------------------------------------
__global__ __launch_bounds__(256)
void finalize_kernel(const float* __restrict__ partial,
                     const float* __restrict__ num,
                     float* __restrict__ out) {
  __shared__ float sh[256];
  const int t = threadIdx.x;
  float local = 0.f;
  for (int i = t; i < BN; i += 256) {
    float rs = 0.f;
#pragma unroll
    for (int js = 0; js < 8 * NWN; ++js) rs += partial[(size_t)js * BN + i];
    float n     = num[i];
    float denom = rs - n;   // off-diagonal row sum
    local += (denom != 0.f) ? -__logf(n / denom) : 0.f;
  }
  sh[t] = local;
  __syncthreads();
  for (int off = 128; off > 0; off >>= 1) {
    if (t < off) sh[t] += sh[t + off];
    __syncthreads();
  }
  if (t == 0) out[0] = sh[0] / (float)BN;
}

// ---------------------------------------------------------------------------
// Workspace layout (bytes):
//   [0,            8 MiB)   Af16 (images, f16)
//   [8 MiB,       16 MiB)   Bf16 (texts,  f16)
//   [16 MiB, +16 KB)        img_sq
//   next 16 KB              txt_sq
//   next 16 KB              num (fp32 diagonal sim)
//   next 512 KB             partial row sums [8*NWN][4096]
// Total ~16.6 MiB.
// ---------------------------------------------------------------------------
extern "C" void kernel_launch(void* const* d_in, const int* in_sizes, int n_in,
                              void* d_out, int out_size, void* d_ws, size_t ws_size,
                              hipStream_t stream) {
  const float* txt = (const float*)d_in[0];   // text_embeddings
  const float* img = (const float*)d_in[1];   // image_embeddings

  char* ws = (char*)d_ws;
  _Float16* Af16  = (_Float16*)(ws);
  _Float16* Bf16  = (_Float16*)(ws + (size_t)8 * 1024 * 1024);
  float* img_sq   = (float*)(ws + (size_t)16 * 1024 * 1024);
  float* txt_sq   = img_sq + BN;
  float* num      = txt_sq + BN;
  float* partial  = num + BN;
  float* out      = (float*)d_out;

  prep_kernel<<<BN, 256, 0, stream>>>(txt, img, Af16, Bf16, img_sq, txt_sq, num);

  dim3 grid(BN / 128, 8);
  simrow_kernel<<<grid, 256, 0, stream>>>(Af16, Bf16, img_sq, txt_sq, partial);

  finalize_kernel<<<1, 256, 0, stream>>>(partial, num, out);
}
